// ANIModel_25383256719857
// MI455X (gfx1250) — matrix operand order repaired and checked by reading, not compiled, not run on hardware
//
#include <hip/hip_runtime.h>
#include <math.h>

// ---------------------------------------------------------------------------
// ANI per-species MLP on MI455X (gfx1250), f16 WMMA path.
//   AEV traffic 201MB @23.3TB/s ~ 8.6us ; 24.7 GFLOP selected-expert compute.
//   Bucket-by-species so each 16x16 WMMA tile uses a single expert's weights.
//   v2: double-buffered L1 weight slices (copy overlaps WMMA, 1 barrier/iter),
//       B-tile register rotation so WMMA waits on dscnt<=2 instead of 0,
//       global_prefetch of the next weight slice.
// ---------------------------------------------------------------------------

typedef __attribute__((ext_vector_type(16))) _Float16 v16h;
typedef __attribute__((ext_vector_type(8)))  float    v8f;

struct __attribute__((aligned(16))) U4 { unsigned int x, y, z, w; };
union V16 { v16h h; U4 q[2]; };

#define NWAVE 4   // waves per block, each owns one 16-atom tile

__device__ inline v8f wmma_f16(v16h a, v16h b, v8f c) {
  // v_wmma_f32_16x16x32_f16 : D = A(16x32,f16) x B(32x16,f16) + C(16x16,f32)
  return __builtin_amdgcn_wmma_f32_16x16x32_f16(false, a, false, b,
                                                (short)0, c, false, false);
}

// A-matrix lane layout (16-bit, 16x32): lane = (m, hi); VGPR j holds f16 pair
//   j<4 : K = kb*32 + 2j + 8*hi ; j>=4 : K = kb*32 + 16 + 2(j-4) + 8*hi
// => dwords [kb*16+4hi .. +3] and [kb*16+4hi+8 .. +3] of row m.
__device__ inline v16h load_tileA(const _Float16* row, int kb, int hi) {
  const U4* p = (const U4*)row;           // row is 16B aligned by construction
  V16 r;
  r.q[0] = p[kb * 4 + hi];
  r.q[1] = p[kb * 4 + hi + 2];
  return r.h;
}

// B-matrix lane layout (16-bit, 32x16): lane = (n, hi); same K order as A
//   => dwords [kb*16+4hi .. +3] and [kb*16+4hi+8 .. +3] of W^T row n.
__device__ inline v16h load_tileB(const _Float16* mat, int rowIdx,
                                  int strideHalves, int kb, int hi) {
  const U4* p = (const U4*)(mat + rowIdx * strideHalves);
  V16 r;
  r.q[0] = p[kb * 4 + hi];
  r.q[1] = p[kb * 4 + hi + 2];
  return r.h;
}

__device__ inline float celu01(float x) {
  return x > 0.f ? x : 0.1f * (__expf(x * 10.f) - 1.f);
}

// --------------------------- bucketing kernels -----------------------------

__global__ void ani_zero_small(int* counts, int* cursors) {
  int t = threadIdx.x;
  if (t < 4) { counts[t] = 0; cursors[t] = 0; }
}

__global__ void ani_count(const int* __restrict__ species, int* counts,
                          float* __restrict__ out_species, int N) {
  int i = blockIdx.x * blockDim.x + threadIdx.x;
  if (i < N) {
    int s = species[i];
    atomicAdd(&counts[s], 1);
    out_species[i] = (float)s;   // passthrough output #0
  }
}

__global__ void ani_scan(const int* counts, int* offsets) {
  if (threadIdx.x == 0) {
    int acc = 0;
    for (int s = 0; s < 4; ++s) { offsets[s] = acc; acc += counts[s]; }
    offsets[4] = acc;
  }
}

__global__ void ani_place(const int* __restrict__ species, int* cursors,
                          const int* __restrict__ offsets, int* idxbuf, int N) {
  int i = blockIdx.x * blockDim.x + threadIdx.x;
  if (i < N) {
    int s = species[i];
    int p = atomicAdd(&cursors[s], 1);
    idxbuf[offsets[s] + p] = i;  // order within bucket irrelevant to results
  }
}

// W[e][din][dout] f32  ->  WT[e][dout][din] f16 (transposed for B-tile loads)
__global__ void ani_convert_w(const float* __restrict__ W,
                              _Float16* __restrict__ WT,
                              int din, int dout, int total) {
  int i = blockIdx.x * blockDim.x + threadIdx.x;
  if (i >= total) return;
  int per = din * dout;
  int e = i / per, r = i - e * per;
  int d = r / dout, o = r - d * dout;
  WT[e * per + o * din + d] = (_Float16)W[i];
}

// ------------------------------ main MLP -----------------------------------

__global__ __launch_bounds__(32 * NWAVE, 1)
void ani_mlp(const float* __restrict__ aev,
             const float* __restrict__ b1, const float* __restrict__ b2,
             const float* __restrict__ b3, const float* __restrict__ b4,
             const _Float16* __restrict__ wt1, const _Float16* __restrict__ wt2,
             const _Float16* __restrict__ wt3, const _Float16* __restrict__ wt4,
             const int* __restrict__ counts, const int* __restrict__ offsets,
             const int* __restrict__ idxbuf, float* __restrict__ eatom) {
  // LDS: 40960 + 24576 + 2*10240 + 49152 + 20480 = 155648 B
  __shared__ __attribute__((aligned(16))) _Float16 sW2[128 * 160];
  __shared__ __attribute__((aligned(16))) _Float16 sW3[96 * 128];
  __shared__ __attribute__((aligned(16))) _Float16 sChunk[2][160 * 32];
  __shared__ __attribute__((aligned(16))) _Float16 sActA[NWAVE][16 * 384];
  __shared__ __attribute__((aligned(16))) _Float16 sActB[NWAVE][16 * 160];

  const int s = blockIdx.y;
  const int cnt = counts[s];
  const int tiles = (cnt + 15) >> 4;
  if ((int)(blockIdx.x * NWAVE) >= tiles) return;   // uniform early exit
  const int off = offsets[s];

  const int tid = threadIdx.x;
  const int wave = tid >> 5, lane = tid & 31;
  const int m = lane & 15, hi = lane >> 4;

  // Stage this expert's layer-2/3 weights into LDS (shared by all waves).
  {
    const U4* src2 = (const U4*)(wt2 + (size_t)s * 128 * 160);
    U4* dst2 = (U4*)sW2;
    for (int i = tid; i < 128 * 160 / 8; i += 32 * NWAVE) dst2[i] = src2[i];
    const U4* src3 = (const U4*)(wt3 + (size_t)s * 96 * 128);
    U4* dst3 = (U4*)sW3;
    for (int i = tid; i < 96 * 128 / 8; i += 32 * NWAVE) dst3[i] = src3[i];
  }

  const int t = blockIdx.x * NWAVE + wave;
  const bool wvalid = t < tiles;
  const int tt = wvalid ? t : 0;      // invalid waves replay tile 0 (masked out)

  // Stage 16 gathered AEV rows (f32 -> f16) into this wave's A buffer.
  {
    _Float16* dst = sActA[wave];
    for (int q = lane; q < 16 * 96; q += 32) {
      int r = q / 96, c = q - r * 96;               // c in float4 units
      int ai = tt * 16 + r;
      int ga = (ai < cnt) ? idxbuf[off + ai] : idxbuf[off];
      float4 v = ((const float4*)(aev + (size_t)ga * 384))[c];
      _Float16* d = dst + r * 384 + c * 4;
      d[0] = (_Float16)v.x; d[1] = (_Float16)v.y;
      d[2] = (_Float16)v.z; d[3] = (_Float16)v.w;
    }
  }

  v8f vz = {0.f, 0.f, 0.f, 0.f, 0.f, 0.f, 0.f, 0.f};

  // ---- Layer 1: 384 -> 160 (weights streamed in double-buffered K-slices) ----
  v8f acc1[10];
#pragma unroll
  for (int nb = 0; nb < 10; ++nb) acc1[nb] = vz;

  const _Float16* aRow1 = sActA[wave] + m * 384;
  const U4* wt1r = (const U4*)(wt1 + (size_t)s * 160 * 384);  // row = 48 U4

  // stage slice kb into buffer buf; prefetch slice kb+1 while at it
  auto stage_chunk = [&](int kb, int buf) {
    U4* cd = (U4*)sChunk[buf];
    for (int i = tid; i < 640; i += 32 * NWAVE) {
      int n = i >> 2, j = i & 3;
      cd[i] = wt1r[n * 48 + kb * 4 + j];
      if (kb + 1 < 12)
        __builtin_prefetch((const void*)(wt1r + n * 48 + (kb + 1) * 4 + j), 0, 0);
    }
  };

  stage_chunk(0, 0);
  for (int kb = 0; kb < 12; ++kb) {
    __syncthreads();                 // slice kb&1 ready; slice (kb+1)&1 free
    if (kb + 1 < 12) stage_chunk(kb + 1, (kb + 1) & 1);   // overlaps WMMAs
    v16h a = load_tileA(aRow1, kb, hi);
    const _Float16* ch = sChunk[kb & 1];
    v16h bcur = load_tileB(ch, m, 32, 0, hi);
#pragma unroll
    for (int nb = 0; nb < 10; ++nb) {
      v16h bnxt = load_tileB(ch, ((nb + 1 < 10) ? nb + 1 : nb) * 16 + m, 32, 0, hi);
      acc1[nb] = wmma_f16(a, bcur, acc1[nb]);
      bcur = bnxt;                   // next B in flight during this WMMA
    }
  }
  {  // bias + CELU, f32 D-tile -> f16 activations (stride 160)
    _Float16* dst = sActB[wave];
#pragma unroll
    for (int nb = 0; nb < 10; ++nb) {
      float bias = b1[s * 160 + nb * 16 + m];
#pragma unroll
      for (int r = 0; r < 8; ++r) {
        float y = celu01(acc1[nb][r] + bias);
        dst[(r + 8 * hi) * 160 + nb * 16 + m] = (_Float16)y;  // D: col=n, row=r+8hi
      }
    }
  }
  __syncthreads();

  // ---- Layer 2: 160 -> 128 ----
  v8f acc2[8];
#pragma unroll
  for (int nb = 0; nb < 8; ++nb) acc2[nb] = vz;
  const _Float16* aRow2 = sActB[wave] + m * 160;
#pragma unroll
  for (int kb = 0; kb < 5; ++kb) {
    v16h a = load_tileA(aRow2, kb, hi);
    v16h bcur = load_tileB(sW2, m, 160, kb, hi);
#pragma unroll
    for (int nb = 0; nb < 8; ++nb) {
      v16h bnxt = load_tileB(sW2, ((nb + 1 < 8) ? nb + 1 : nb) * 16 + m, 160, kb, hi);
      acc2[nb] = wmma_f16(a, bcur, acc2[nb]);
      bcur = bnxt;
    }
  }
  {
    _Float16* dst = sActA[wave];           // reuse as h2, stride 128
#pragma unroll
    for (int nb = 0; nb < 8; ++nb) {
      float bias = b2[s * 128 + nb * 16 + m];
#pragma unroll
      for (int r = 0; r < 8; ++r) {
        float y = celu01(acc2[nb][r] + bias);
        dst[(r + 8 * hi) * 128 + nb * 16 + m] = (_Float16)y;
      }
    }
  }
  __syncthreads();

  // ---- Layer 3: 128 -> 96 ----
  v8f acc3[6];
#pragma unroll
  for (int nb = 0; nb < 6; ++nb) acc3[nb] = vz;
  const _Float16* aRow3 = sActA[wave] + m * 128;
#pragma unroll
  for (int kb = 0; kb < 4; ++kb) {
    v16h a = load_tileA(aRow3, kb, hi);
    v16h bcur = load_tileB(sW3, m, 128, kb, hi);
#pragma unroll
    for (int nb = 0; nb < 6; ++nb) {
      v16h bnxt = load_tileB(sW3, ((nb + 1 < 6) ? nb + 1 : nb) * 16 + m, 128, kb, hi);
      acc3[nb] = wmma_f16(a, bcur, acc3[nb]);
      bcur = bnxt;
    }
  }
  {
    _Float16* dst = sActB[wave];           // reuse as h3, stride 96
#pragma unroll
    for (int nb = 0; nb < 6; ++nb) {
      float bias = b3[s * 96 + nb * 16 + m];
#pragma unroll
      for (int r = 0; r < 8; ++r) {
        float y = celu01(acc3[nb][r] + bias);
        dst[(r + 8 * hi) * 96 + nb * 16 + m] = (_Float16)y;
      }
    }
  }
  __syncthreads();

  // ---- Layer 4: 96 -> 1 dot product, lanes 0..15 handle one atom each ----
  if (lane < 16 && wvalid) {
    int ai = t * 16 + lane;
    if (ai < cnt) {
      const _Float16* hrow = sActB[wave] + lane * 96;
      const _Float16* w4 = wt4 + s * 96;
      float acc = 0.f;
      for (int k = 0; k < 96; ++k) acc += (float)hrow[k] * (float)w4[k];
      eatom[idxbuf[off + ai]] = acc + b4[s];
    }
  }
}

// Sum 32 per-atom energies per molecule with a fixed wave32 shuffle tree.
__global__ void ani_reduce(const float* __restrict__ eatom,
                           float* __restrict__ energies, int Bmol) {
  int gid = blockIdx.x * blockDim.x + threadIdx.x;
  int mol = gid >> 5, lane = gid & 31;
  if (mol >= Bmol) return;
  float v = eatom[mol * 32 + lane];
  for (int d = 16; d > 0; d >>= 1) v += __shfl_xor(v, d, 32);
  if (lane == 0) energies[mol] = v;
}

// ------------------------------- host side ---------------------------------

extern "C" void kernel_launch(void* const* d_in, const int* in_sizes, int n_in,
                              void* d_out, int out_size, void* d_ws,
                              size_t ws_size, hipStream_t stream) {
  (void)n_in; (void)out_size; (void)ws_size;
  const int*   species = (const int*)d_in[0];
  const float* aev = (const float*)d_in[1];
  const float* W1 = (const float*)d_in[2];
  const float* b1 = (const float*)d_in[3];
  const float* W2 = (const float*)d_in[4];
  const float* b2 = (const float*)d_in[5];
  const float* W3 = (const float*)d_in[6];
  const float* b3 = (const float*)d_in[7];
  const float* W4 = (const float*)d_in[8];
  const float* b4 = (const float*)d_in[9];

  const int N = in_sizes[0];        // B*A atoms
  const int Bmol = N / 32;
  float* out_species = (float*)d_out;
  float* out_energy  = (float*)d_out + N;

  // workspace layout (~1.72 MB total)
  char* ws = (char*)d_ws;
  int* counts  = (int*)ws;                      // 4
  int* cursors = (int*)(ws + 16);               // 4
  int* offsets = (int*)(ws + 32);               // 5
  int* idxbuf  = (int*)(ws + 256);              // N ints
  float* eatom = (float*)(ws + 256 + (size_t)4 * N);  // N floats
  _Float16* wt1 = (_Float16*)(ws + 256 + (size_t)8 * N);  // 16B aligned
  _Float16* wt2 = wt1 + 4 * 160 * 384;
  _Float16* wt3 = wt2 + 4 * 128 * 160;
  _Float16* wt4 = wt3 + 4 * 96 * 128;

  ani_zero_small<<<1, 32, 0, stream>>>(counts, cursors);
  ani_count<<<(N + 255) / 256, 256, 0, stream>>>(species, counts, out_species, N);
  ani_scan<<<1, 32, 0, stream>>>(counts, offsets);
  ani_place<<<(N + 255) / 256, 256, 0, stream>>>(species, cursors, offsets, idxbuf, N);

  ani_convert_w<<<(245760 + 255) / 256, 256, 0, stream>>>(W1, wt1, 384, 160, 245760);
  ani_convert_w<<<(81920 + 255) / 256, 256, 0, stream>>>(W2, wt2, 160, 128, 81920);
  ani_convert_w<<<(49152 + 255) / 256, 256, 0, stream>>>(W3, wt3, 128, 96, 49152);
  ani_convert_w<<<(384 + 255) / 256, 256, 0, stream>>>(W4, wt4, 96, 1, 384);

  int tilesMax = (N + 15) / 16;                 // worst case: one bucket owns all
  dim3 grid((tilesMax + NWAVE - 1) / NWAVE, 4, 1);
  ani_mlp<<<grid, 32 * NWAVE, 0, stream>>>(aev, b1, b2, b3, b4,
                                           wt1, wt2, wt3, wt4,
                                           counts, offsets, idxbuf, eatom);

  ani_reduce<<<(Bmol * 32 + 255) / 256, 256, 0, stream>>>(eatom, out_energy, Bmol);
}
